// ProductKeyMemory_12137577579026
// MI455X (gfx1250) — compile-verified, matrix operand order checked by reading, not run on hardware
//
#include <hip/hip_runtime.h>
#include <math.h>

typedef __attribute__((ext_vector_type(16))) __bf16 v16bf;
typedef __attribute__((ext_vector_type(8)))  float  v8f;

#define DIMX      1024
#define NSUB      1024
#define KEYD      256
#define CODED     128
#define TOPK      32
#define NTOK      8192   // B*S = 4*2048

// ---------------- helpers ----------------

__device__ __forceinline__ unsigned short f2bf(float f) {
  unsigned int u = __float_as_uint(f);
  u += 0x7fffu + ((u >> 16) & 1u);          // round-to-nearest-even
  return (unsigned short)(u >> 16);
}

__global__ void cvt_f32_to_bf16(const float* __restrict__ in,
                                unsigned short* __restrict__ out, int n) {
  int i = blockIdx.x * 256 + threadIdx.x;
  int stride = gridDim.x * 256;
  for (; i < n; i += stride) out[i] = f2bf(in[i]);
}

// A-frag (16x32 bf16) and B-frag of row-major [N,K] share the same per-lane
// layout: lane L<16 -> row base+L, K 0..15 (32 contiguous bytes);
//         lane L>=16 -> row base+L-16, K 16..31.
__device__ __forceinline__ v16bf load_frag(const unsigned short* __restrict__ p,
                                           int row, int k, int ld, int lane) {
  int r  = row + (lane & 15);
  int kk = k + ((lane >> 4) << 4);
  return *(const v16bf*)(p + (size_t)r * ld + kk);
}

// ---------------- WMMA GEMM: C[M,N] = A[M,K] * B[N,K]^T ----------------
// Block: 256 threads = 8 waves, block tile 64x256.
// Wave tile 32x64: 2 A-frags x 4 B-frags -> 8 WMMAs per K-step (vs 12 b128 loads).
// mode 0: store f32   mode 1: store bf16   mode 2: silu then store bf16
__global__ __launch_bounds__(256) void gemm_bf16_wmma(
    const unsigned short* __restrict__ A, const unsigned short* __restrict__ B,
    void* __restrict__ C, int M, int N, int K,
    int lda, int ldb, int ldc, int mode) {
  const int lane = threadIdx.x & 31;
  const int wave = threadIdx.x >> 5;     // 8 waves (wave32)
  const int wm = wave >> 2;              // 0..1
  const int wn = wave & 3;               // 0..3
  const int m0 = blockIdx.y * 64 + wm * 32;
  const int n0 = blockIdx.x * 256 + wn * 64;
  (void)M; (void)N;

  v8f c[2][4];
  #pragma unroll
  for (int i = 0; i < 2; ++i)
    #pragma unroll
    for (int j = 0; j < 4; ++j) c[i][j] = {};

  for (int k = 0; k < K; k += 32) {
    v16bf a0 = load_frag(A, m0,      k, lda, lane);
    v16bf a1 = load_frag(A, m0 + 16, k, lda, lane);
    v16bf b0 = load_frag(B, n0,      k, ldb, lane);
    v16bf b1 = load_frag(B, n0 + 16, k, ldb, lane);
    v16bf b2 = load_frag(B, n0 + 32, k, ldb, lane);
    v16bf b3 = load_frag(B, n0 + 48, k, ldb, lane);

    // branch-free next-K prefetch (gfx1250 global_prefetch_b8)
    int kp = (k + 32 < K) ? (k + 32) : k;
    __builtin_prefetch(A + (size_t)(m0 + (lane & 15)) * lda + kp, 0, 1);
    __builtin_prefetch(B + (size_t)(n0 + (lane & 15)) * ldb + kp, 0, 1);
    __builtin_prefetch(B + (size_t)(n0 + 32 + (lane & 15)) * ldb + kp, 0, 1);

    c[0][0] = __builtin_amdgcn_wmma_f32_16x16x32_bf16(false, a0, false, b0, (short)0, c[0][0], false, false);
    c[0][1] = __builtin_amdgcn_wmma_f32_16x16x32_bf16(false, a0, false, b1, (short)0, c[0][1], false, false);
    c[0][2] = __builtin_amdgcn_wmma_f32_16x16x32_bf16(false, a0, false, b2, (short)0, c[0][2], false, false);
    c[0][3] = __builtin_amdgcn_wmma_f32_16x16x32_bf16(false, a0, false, b3, (short)0, c[0][3], false, false);
    c[1][0] = __builtin_amdgcn_wmma_f32_16x16x32_bf16(false, a1, false, b0, (short)0, c[1][0], false, false);
    c[1][1] = __builtin_amdgcn_wmma_f32_16x16x32_bf16(false, a1, false, b1, (short)0, c[1][1], false, false);
    c[1][2] = __builtin_amdgcn_wmma_f32_16x16x32_bf16(false, a1, false, b2, (short)0, c[1][2], false, false);
    c[1][3] = __builtin_amdgcn_wmma_f32_16x16x32_bf16(false, a1, false, b3, (short)0, c[1][3], false, false);
  }

  // C/D layout: VGPR r, lanes 0-15 -> (M=r, N=lane); lanes 16-31 -> (M=r+8, N=lane-16)
  const int rbase = (lane >> 4) * 8;
  const int col   = lane & 15;
  float* Cf = (float*)C;
  unsigned short* Cb = (unsigned short*)C;
  for (int fm = 0; fm < 2; ++fm) {
    for (int fn = 0; fn < 4; ++fn) {
      int rowg = m0 + fm * 16 + rbase;
      int colg = n0 + fn * 16 + col;
      #pragma unroll
      for (int r = 0; r < 8; ++r) {
        float v = c[fm][fn][r];
        size_t idx = (size_t)(rowg + r) * ldc + colg;
        if (mode == 0) {
          Cf[idx] = v;
        } else {
          if (mode == 2) v = v / (1.0f + __expf(-v));   // silu
          Cb[idx] = f2bf(v);
        }
      }
    }
  }
}

// ---------------- top-32 of each row (row_len = 1024) ----------------
__global__ __launch_bounds__(256) void topk32_kernel(
    const float* __restrict__ scores, float* __restrict__ ts,
    int* __restrict__ ti, int row_len) {
  const int row = blockIdx.x;
  const int t = threadIdx.x;
  const float* src = scores + (size_t)row * row_len;
  __shared__ float vals[1024];
  __shared__ float rv[256];
  __shared__ int   ri[256];

  for (int i = t; i < row_len; i += 256) vals[i] = src[i];
  __syncthreads();

  for (int k = 0; k < TOPK; ++k) {
    float best = -INFINITY; int bi = row_len;
    for (int i = t; i < row_len; i += 256) {
      float v = vals[i];
      if (v > best || (v == best && i < bi)) { best = v; bi = i; }
    }
    rv[t] = best; ri[t] = bi;
    __syncthreads();
    for (int s = 128; s > 0; s >>= 1) {
      if (t < s) {
        float v = rv[t + s]; int ii = ri[t + s];
        if (v > rv[t] || (v == rv[t] && ii < ri[t])) { rv[t] = v; ri[t] = ii; }
      }
      __syncthreads();
    }
    if (t == 0) {
      ts[(size_t)row * TOPK + k] = rv[0];
      ti[(size_t)row * TOPK + k] = ri[0];
      vals[ri[0]] = -INFINITY;
    }
    __syncthreads();
  }
}

// ------- candidate top-32, softmax, weighted code gather -> mixed (bf16) -------
__global__ __launch_bounds__(256) void cand_mix_kernel(
    const float* __restrict__ tas, const int* __restrict__ tai,
    const float* __restrict__ tbs, const int* __restrict__ tbi,
    const float* __restrict__ codes, unsigned short* __restrict__ mixed) {
  const int tok = blockIdx.x;
  const int t = threadIdx.x;
  __shared__ float as_[TOPK], bs_[TOPK];
  __shared__ int   ai_[TOPK], bi_[TOPK];
  __shared__ float cand[1024];
  __shared__ float rv[256];
  __shared__ int   ri[256];
  __shared__ float w[TOPK];
  __shared__ int   fidx[TOPK];

  if (t < TOPK) {
    as_[t] = tas[(size_t)tok * TOPK + t];
    ai_[t] = tai[(size_t)tok * TOPK + t];
    bs_[t] = tbs[(size_t)tok * TOPK + t];
    bi_[t] = tbi[(size_t)tok * TOPK + t];
  }
  __syncthreads();
  for (int i = t; i < 1024; i += 256) cand[i] = as_[i >> 5] + bs_[i & 31];
  __syncthreads();

  for (int k = 0; k < TOPK; ++k) {
    float best = -INFINITY; int bi = 1024;
    for (int i = t; i < 1024; i += 256) {
      float v = cand[i];
      if (v > best || (v == best && i < bi)) { best = v; bi = i; }
    }
    rv[t] = best; ri[t] = bi;
    __syncthreads();
    for (int s = 128; s > 0; s >>= 1) {
      if (t < s) {
        float v = rv[t + s]; int ii = ri[t + s];
        if (v > rv[t] || (v == rv[t] && ii < ri[t])) { rv[t] = v; ri[t] = ii; }
      }
      __syncthreads();
    }
    if (t == 0) {
      int p = ri[0];
      w[k]    = rv[0];
      fidx[k] = ai_[p >> 5] * NSUB + bi_[p & 31];
      cand[p] = -INFINITY;
    }
    __syncthreads();
  }

  if (t == 0) {  // softmax(final_scores / sqrt(2*KEY_DIM))
    const float inv = 0.04419417382415922f;   // 1/sqrt(512)
    float m = w[0];
    for (int k = 1; k < TOPK; ++k) m = fmaxf(m, w[k]);
    float sum = 0.0f;
    for (int k = 0; k < TOPK; ++k) { float e = __expf((w[k] - m) * inv); w[k] = e; sum += e; }
    float r = 1.0f / sum;
    for (int k = 0; k < TOPK; ++k) w[k] *= r;
  }
  __syncthreads();

  if (t < CODED) {
    float acc = 0.0f;
    #pragma unroll 4
    for (int k = 0; k < TOPK; ++k)
      acc += w[k] * codes[(size_t)fidx[k] * CODED + t];
    mixed[(size_t)tok * CODED + t] = f2bf(acc);
  }
}

// ---------------- launcher ----------------
extern "C" void kernel_launch(void* const* d_in, const int* in_sizes, int n_in,
                              void* d_out, int out_size, void* d_ws, size_t ws_size,
                              hipStream_t stream) {
  (void)in_sizes; (void)n_in; (void)out_size; (void)ws_size;
  const float* x     = (const float*)d_in[0];
  const float* Wq    = (const float*)d_in[1];
  const float* key_a = (const float*)d_in[2];
  const float* key_b = (const float*)d_in[3];
  const float* codes = (const float*)d_in[4];
  const float* W1    = (const float*)d_in[5];
  const float* W2    = (const float*)d_in[6];
  float* out = (float*)d_out;

  char* ws = (char*)d_ws;
  size_t off = 0;
  auto alloc = [&](size_t bytes) -> void* {
    void* p = ws + off;
    off = (off + bytes + 255) & ~(size_t)255;
    return p;
  };

  unsigned short* xb  = (unsigned short*)alloc((size_t)NTOK * DIMX * 2);
  unsigned short* wqb = (unsigned short*)alloc((size_t)(2 * KEYD) * DIMX * 2);
  unsigned short* kab = (unsigned short*)alloc((size_t)NSUB * KEYD * 2);
  unsigned short* kbb = (unsigned short*)alloc((size_t)NSUB * KEYD * 2);
  unsigned short* w1b = (unsigned short*)alloc((size_t)DIMX * CODED * 2);
  unsigned short* w2b = (unsigned short*)alloc((size_t)DIMX * DIMX * 2);
  unsigned short* qb  = (unsigned short*)alloc((size_t)NTOK * (2 * KEYD) * 2);
  float* sa  = (float*)alloc((size_t)NTOK * NSUB * 4);
  float* sb  = (float*)alloc((size_t)NTOK * NSUB * 4);
  float* tas = (float*)alloc((size_t)NTOK * TOPK * 4);
  int*   tai = (int*)  alloc((size_t)NTOK * TOPK * 4);
  float* tbs = (float*)alloc((size_t)NTOK * TOPK * 4);
  int*   tbi = (int*)  alloc((size_t)NTOK * TOPK * 4);
  unsigned short* mixedb = (unsigned short*)alloc((size_t)NTOK * CODED * 2);
  unsigned short* hb     = (unsigned short*)alloc((size_t)NTOK * DIMX * 2);

  auto cvt = [&](const float* src, unsigned short* dst, int n) {
    int blocks = (n + 255) / 256;
    if (blocks > 16384) blocks = 16384;
    cvt_f32_to_bf16<<<blocks, 256, 0, stream>>>(src, dst, n);
  };
  cvt(x,     xb,  NTOK * DIMX);
  cvt(Wq,    wqb, 2 * KEYD * DIMX);
  cvt(key_a, kab, NSUB * KEYD);
  cvt(key_b, kbb, NSUB * KEYD);
  cvt(W1,    w1b, DIMX * CODED);
  cvt(W2,    w2b, DIMX * DIMX);

  // q = x @ Wq^T  -> bf16 [NTOK, 512]
  gemm_bf16_wmma<<<dim3((2 * KEYD) / 256, NTOK / 64), 256, 0, stream>>>(
      xb, wqb, qb, NTOK, 2 * KEYD, DIMX, DIMX, DIMX, 2 * KEYD, 1);

  // scores_a = q[:, :256] @ key_a^T ; scores_b = q[:, 256:] @ key_b^T  (f32)
  gemm_bf16_wmma<<<dim3(NSUB / 256, NTOK / 64), 256, 0, stream>>>(
      qb,        kab, sa, NTOK, NSUB, KEYD, 2 * KEYD, KEYD, NSUB, 0);
  gemm_bf16_wmma<<<dim3(NSUB / 256, NTOK / 64), 256, 0, stream>>>(
      qb + KEYD, kbb, sb, NTOK, NSUB, KEYD, 2 * KEYD, KEYD, NSUB, 0);

  // per-row top-32
  topk32_kernel<<<NTOK, 256, 0, stream>>>(sa, tas, tai, NSUB);
  topk32_kernel<<<NTOK, 256, 0, stream>>>(sb, tbs, tbi, NSUB);

  // candidate top-32 + softmax + code mix -> mixed bf16 [NTOK, 128]
  cand_mix_kernel<<<NTOK, 256, 0, stream>>>(tas, tai, tbs, tbi, codes, mixedb);

  // h = silu(mixed @ W1^T) -> bf16 [NTOK, 1024]
  gemm_bf16_wmma<<<dim3(DIMX / 256, NTOK / 64), 256, 0, stream>>>(
      mixedb, w1b, hb, NTOK, DIMX, CODED, CODED, CODED, DIMX, 2);

  // out = h @ W2^T -> f32 [NTOK, 1024]
  gemm_bf16_wmma<<<dim3(DIMX / 256, NTOK / 64), 256, 0, stream>>>(
      hb, w2b, out, NTOK, DIMX, DIMX, DIMX, DIMX, DIMX, 0);
}
